// Perceiver_62457414418852
// MI455X (gfx1250) — compile-verified, matrix operand order checked by reading, not run on hardware
//
#include <hip/hip_runtime.h>
#include <hip/hip_bf16.h>
#include <math.h>

// ---------------------------------------------------------------------------
// Perceiver forward for MI455X (gfx1250, wave32, WMMA).
// All GEMMs run through v_wmma_f32_16x16x32_bf16 (fp32 accumulate, bf16
// operands) with fully branchless fragment loads (float4 vector loads for A
// and transposed-B; clamped coalesced loads for row-major B).
// Input pointer order assumption (JAX pytree flatten: top-level dict
// insertion order, nested dicts sorted by key, lists in order):
//   0 input  1 query_color  2 mask(ignored; all-True)
//   3..10  ca1  {k.b,k.w,o.b,o.w,q.b,q.w,v.b,v.w}
//   11..18 ca2  (same)
//   19,20  color {b,w}
//   21..28 dec  {k.b,k.w,o.b,o.w,q.b,q.w,v.b,v.w}
//   29     latent_init
//   30+16i layers[i] {ffn1.b,ffn1.w,ffn2.b,ffn2.w,ln1b,ln1g,ln2b,ln2g,
//                     mha.k.b,mha.k.w,mha.o.b,mha.o.w,mha.q.b,mha.q.w,
//                     mha.v.b,mha.v.w}
//   94,95  sigma {b,w}
// Output: color(8*1024*3) | sigma(8) | latent(8*256*512) | attn(8*256*2048)
// ---------------------------------------------------------------------------

typedef __attribute__((ext_vector_type(16))) __bf16 v16bf;
typedef __attribute__((ext_vector_type(8)))  float  v8f;

#define PB  8
#define PS  2048
#define PQ  1024
#define PD  512
#define PNL 256
#define PH  8
#define PDD 32
#define PDC 64

__device__ __forceinline__ v8f wmma_bf16(v16bf a, v16bf b, v8f c) {
  // (neg_a, A, neg_b, B, c_mod, C, reuse_a, reuse_b)
  return __builtin_amdgcn_wmma_f32_16x16x32_bf16(false, a, false, b, (short)0, c,
                                                 false, false);
}

// A-style fragment (16x32 bf16), branchless. Lane holds row r0+(lane&15);
// half h=(lane>>4) selects K-group. Each lane reads two contiguous 8-float
// segments: [k0+8h .. +7] and [k0+16+8h .. +7]  ->  4x global_load_b128.
// Caller guarantees rows and K-range are in bounds (all dims padded/aligned).
__device__ __forceinline__ v16bf frag_rowK(const float* __restrict__ P, int ld,
                                           int r0, int k0, int lane) {
  int r = r0 + (lane & 15);
  int h = lane >> 4;
  const float* base = P + (long long)r * ld + k0 + (h << 3);
  float4 x0 = *(const float4*)(base + 0);
  float4 x1 = *(const float4*)(base + 4);
  float4 y0 = *(const float4*)(base + 16);
  float4 y1 = *(const float4*)(base + 20);
  v16bf f;
  f[0]  = (__bf16)x0.x; f[1]  = (__bf16)x0.y;
  f[2]  = (__bf16)x0.z; f[3]  = (__bf16)x0.w;
  f[4]  = (__bf16)x1.x; f[5]  = (__bf16)x1.y;
  f[6]  = (__bf16)x1.z; f[7]  = (__bf16)x1.w;
  f[8]  = (__bf16)y0.x; f[9]  = (__bf16)y0.y;
  f[10] = (__bf16)y0.z; f[11] = (__bf16)y0.w;
  f[12] = (__bf16)y1.x; f[13] = (__bf16)y1.y;
  f[14] = (__bf16)y1.z; f[15] = (__bf16)y1.w;
  return f;
}

// B (KxN row-major) fragment, branchless. Lane holds column n0+(lane&15);
// n is clamped (v_min) so out-of-range columns read valid memory; their
// results are discarded by the store guard. Per fixed K the 16 lanes read
// 16 consecutive floats -> coalesced.
__device__ __forceinline__ v16bf frag_colN(const float* __restrict__ P, int ld,
                                           int n0, int N, int k0, int lane) {
  int n = n0 + (lane & 15);
  n = (n < N) ? n : (N - 1);
  int h = lane >> 4;
  const float* c0 = P + n + (long long)(k0 + (h << 3)) * ld;
  const float* c1 = c0 + (long long)16 * ld;
  v16bf f;
#pragma unroll
  for (int j = 0; j < 4; ++j) {
    f[2 * j]     = (__bf16)c0[(long long)(2 * j) * ld];
    f[2 * j + 1] = (__bf16)c0[(long long)(2 * j + 1) * ld];
    f[8 + 2 * j] = (__bf16)c1[(long long)(2 * j) * ld];
    f[9 + 2 * j] = (__bf16)c1[(long long)(2 * j + 1) * ld];
  }
  return f;
}

__device__ __forceinline__ void epi_store(float* __restrict__ C, int ldc, int N,
                                          int m, int n, float v,
                                          const float* __restrict__ bias,
                                          float alpha, int relu) {
  if (n < N) {  // m always in range (M % 128 == 0); only N=32 GEMMs clip
    v *= alpha;
    if (bias) v += bias[n];
    if (relu) v = fmaxf(v, 0.f);
    C[(long long)m * ldc + n] = v;
  }
}

// Batched GEMM: C[b,h] = alpha * A[b,h] x op(B[b,h]) (+bias) (relu).
// BT=false: B is [K,N] row-major. BT=true: B is [N,K] row-major (Q*K^T).
// Block = 8 waves; wave tile 32x32 (2x2 WMMA frags); block tile 128(M)x64(N).
// Requirements (all satisfied by this network): M%128==0, K%32==0; for
// BT=true additionally N%64==0.
template <bool BT>
__global__ __launch_bounds__(256, 2) void k_wmma_gemm(
    const float* __restrict__ A, const float* __restrict__ B,
    const float* __restrict__ bias, float* __restrict__ C, int M, int N, int K,
    int lda, int ldb, int ldc, int nh, long long sAb, long long sAh,
    long long sBb, long long sBh, long long sCb, long long sCh, float alpha,
    int relu_out) {
  int z  = blockIdx.z;
  int ib = z / nh, ih = z % nh;
  A += (long long)ib * sAb + (long long)ih * sAh;
  B += (long long)ib * sBb + (long long)ih * sBh;
  C += (long long)ib * sCb + (long long)ih * sCh;

  int lane = threadIdx.x & 31;
  int wv   = threadIdx.x >> 5;
  int m0   = blockIdx.x * 128 + (wv >> 1) * 32;
  int n0   = blockIdx.y * 64 + (wv & 1) * 32;

  v8f acc00 = {}, acc01 = {}, acc10 = {}, acc11 = {};
  for (int k0 = 0; k0 < K; k0 += 32) {
    v16bf a0 = frag_rowK(A, lda, m0, k0, lane);
    v16bf a1 = frag_rowK(A, lda, m0 + 16, k0, lane);
    v16bf b0, b1;
    if (BT) {
      b0 = frag_rowK(B, ldb, n0, k0, lane);
      b1 = frag_rowK(B, ldb, n0 + 16, k0, lane);
    } else {
      b0 = frag_colN(B, ldb, n0, N, k0, lane);
      b1 = frag_colN(B, ldb, n0 + 16, N, k0, lane);
    }
    acc00 = wmma_bf16(a0, b0, acc00);
    acc01 = wmma_bf16(a0, b1, acc01);
    acc10 = wmma_bf16(a1, b0, acc10);
    acc11 = wmma_bf16(a1, b1, acc11);
  }

  // C/D layout: lane -> n = n0+(lane&15); VGPR r -> m = m0 + r + 8*(lane>>4)
  int nc = n0 + (lane & 15);
  int mh = (lane >> 4) << 3;
#pragma unroll
  for (int r = 0; r < 8; ++r) {
    int mA = m0 + r + mh;
    epi_store(C, ldc, N, mA,      nc,      acc00[r], bias, alpha, relu_out);
    epi_store(C, ldc, N, mA,      nc + 16, acc01[r], bias, alpha, relu_out);
    epi_store(C, ldc, N, mA + 16, nc,      acc10[r], bias, alpha, relu_out);
    epi_store(C, ldc, N, mA + 16, nc + 16, acc11[r], bias, alpha, relu_out);
  }
}

// Row softmax, one block (256 thr) per row.
__global__ void k_softmax(float* __restrict__ S, int cols) {
  float* row = S + (long long)blockIdx.x * cols;
  __shared__ float red[256];
  float m = -INFINITY;
  for (int c = threadIdx.x; c < cols; c += 256) m = fmaxf(m, row[c]);
  red[threadIdx.x] = m;
  __syncthreads();
  for (int o = 128; o > 0; o >>= 1) {
    if (threadIdx.x < o) red[threadIdx.x] = fmaxf(red[threadIdx.x], red[threadIdx.x + o]);
    __syncthreads();
  }
  m = red[0];
  __syncthreads();
  float sum = 0.f;
  for (int c = threadIdx.x; c < cols; c += 256) {
    float e = __expf(row[c] - m);
    row[c]  = e;
    sum += e;
  }
  red[threadIdx.x] = sum;
  __syncthreads();
  for (int o = 128; o > 0; o >>= 1) {
    if (threadIdx.x < o) red[threadIdx.x] += red[threadIdx.x + o];
    __syncthreads();
  }
  float inv = 1.0f / red[0];
  for (int c = threadIdx.x; c < cols; c += 256) row[c] *= inv;
}

// x = LayerNorm(a + x) * g + b ; one block per row
__global__ void k_ln_res(const float* __restrict__ a, float* __restrict__ x,
                         const float* __restrict__ g, const float* __restrict__ b,
                         int d) {
  long long row = blockIdx.x;
  const float* ar = a + row * d;
  float* xr = x + row * d;
  __shared__ float red[256];
  float s = 0.f, ss = 0.f;
  for (int c = threadIdx.x; c < d; c += 256) {
    float v = ar[c] + xr[c];
    s += v;
    ss += v * v;
  }
  red[threadIdx.x] = s;
  __syncthreads();
  for (int o = 128; o > 0; o >>= 1) {
    if (threadIdx.x < o) red[threadIdx.x] += red[threadIdx.x + o];
    __syncthreads();
  }
  float mean = red[0] / d;
  __syncthreads();
  red[threadIdx.x] = ss;
  __syncthreads();
  for (int o = 128; o > 0; o >>= 1) {
    if (threadIdx.x < o) red[threadIdx.x] += red[threadIdx.x + o];
    __syncthreads();
  }
  float var = red[0] / d - mean * mean;
  float inv = rsqrtf(var + 1e-5f);
  for (int c = threadIdx.x; c < d; c += 256) {
    float v = ar[c] + xr[c];
    xr[c]   = (v - mean) * inv * g[c] + b[c];
  }
}

__global__ void k_relu(float* __restrict__ x, long long n) {
  long long i  = (long long)blockIdx.x * blockDim.x + threadIdx.x;
  long long st = (long long)gridDim.x * blockDim.x;
  for (; i < n; i += st) x[i] = fmaxf(x[i], 0.f);
}

__global__ void k_bcast(const float* __restrict__ src, float* __restrict__ dst,
                        long long per, long long total) {
  long long i  = (long long)blockIdx.x * blockDim.x + threadIdx.x;
  long long st = (long long)gridDim.x * blockDim.x;
  for (; i < total; i += st) dst[i] = src[i % per];
}

// sigma[b] = sum_d relu(max_n latent[b,n,d]) * w[d] + bias
__global__ void k_sigma(const float* __restrict__ lat, const float* __restrict__ w,
                        const float* __restrict__ bias, float* __restrict__ out) {
  int b = blockIdx.x;
  __shared__ float red[256];
  float acc = 0.f;
  for (int d = threadIdx.x; d < PD; d += 256) {
    const float* p = lat + (long long)b * PNL * PD + d;
    float m = -INFINITY;
    for (int n = 0; n < PNL; ++n) m = fmaxf(m, p[(long long)n * PD]);
    acc += fmaxf(m, 0.f) * w[d];
  }
  red[threadIdx.x] = acc;
  __syncthreads();
  for (int o = 128; o > 0; o >>= 1) {
    if (threadIdx.x < o) red[threadIdx.x] += red[threadIdx.x + o];
    __syncthreads();
  }
  if (threadIdx.x == 0) out[b] = red[0] + bias[0];
}

// color head: out[m,0..2] = X[m,:] @ W[32,3] + b   (X already relu'ed)
__global__ void k_color(const float* __restrict__ X, const float* __restrict__ W,
                        const float* __restrict__ b, float* __restrict__ out) {
  int m = blockIdx.x * blockDim.x + threadIdx.x;
  if (m >= PB * PQ) return;
  const float* x = X + (long long)m * PDD;
  float c0 = b[0], c1 = b[1], c2 = b[2];
#pragma unroll
  for (int k = 0; k < PDD; ++k) {
    float xv = x[k];
    c0 += xv * W[k * 3 + 0];
    c1 += xv * W[k * 3 + 1];
    c2 += xv * W[k * 3 + 2];
  }
  out[(long long)m * 3 + 0] = c0;
  out[(long long)m * 3 + 1] = c1;
  out[(long long)m * 3 + 2] = c2;
}

// ------------------------------- host side ---------------------------------

static inline void gemm(hipStream_t st, bool bt, const float* A, const float* B,
                        const float* bias, float* C, int M, int N, int K,
                        int lda, int ldb, int ldc, int nb, long long sAb,
                        long long sBb, long long sCb, int nh, long long sAh,
                        long long sBh, long long sCh, float alpha, bool relu) {
  dim3 grid((M + 127) / 128, (N + 63) / 64, nb * nh);
  dim3 block(256);
  if (bt)
    k_wmma_gemm<true><<<grid, block, 0, st>>>(A, B, bias, C, M, N, K, lda, ldb,
                                              ldc, nh, sAb, sAh, sBb, sBh, sCb,
                                              sCh, alpha, relu ? 1 : 0);
  else
    k_wmma_gemm<false><<<grid, block, 0, st>>>(A, B, bias, C, M, N, K, lda, ldb,
                                               ldc, nh, sAb, sAh, sBb, sBh, sCb,
                                               sCh, alpha, relu ? 1 : 0);
}

static inline void relu_ip(hipStream_t st, float* x, long long n) {
  int g = (int)((n + 255) / 256);
  if (g > 4096) g = 4096;
  k_relu<<<g, 256, 0, st>>>(x, n);
}

extern "C" void kernel_launch(void* const* d_in, const int* in_sizes, int n_in,
                              void* d_out, int out_size, void* d_ws,
                              size_t ws_size, hipStream_t stream) {
  auto F = [&](int i) { return (const float*)d_in[i]; };
  (void)in_sizes; (void)n_in; (void)out_size; (void)ws_size;

  const float* input = F(0);
  const float* query = F(1);
  // F(2) = mask, all-True -> softmax unmasked.

  // ca1 / ca2 (sorted keys: k,o,q,v each {b,w})
  const float *c1kb = F(3),  *c1kw = F(4),  *c1ob = F(5),  *c1ow = F(6);
  const float *c1qb = F(7),  *c1qw = F(8),  *c1vb = F(9),  *c1vw = F(10);
  const float *c2kb = F(11), *c2kw = F(12), *c2ob = F(13), *c2ow = F(14);
  const float *c2qb = F(15), *c2qw = F(16), *c2vb = F(17), *c2vw = F(18);
  const float *colb = F(19), *colw = F(20);
  const float *dkb = F(21), *dkw = F(22), *dob = F(23), *dow = F(24);
  const float *dqb = F(25), *dqw = F(26), *dvb = F(27), *dvw = F(28);
  const float* lat_init = F(29);
  const int LBASE = 30;
  const float *sigb = F(94), *sigw = F(95);

  // Output regions
  float* out       = (float*)d_out;
  float* out_color = out;                         // 8*1024*3
  float* out_sigma = out + PB * PQ * 3;           // 8
  float* out_lat   = out_sigma + PB;              // 8*256*512  (live latent)
  float* out_attn  = out_lat + PB * PNL * PD;     // 8*256*2048

  // Workspace carving (floats)
  float* ws   = (float*)d_ws;
  const long long BIG = 8388608LL;  // 2048*4096 == 16384*512
  float* BUF0 = ws;                 // cross-K / enc-Q / dec-q
  float* BUF1 = ws + BIG;           // cross-V / enc-K / dec-k
  float* BUF2 = ws + 2 * BIG;       // enc-V / dec-v
  float* BUF3 = ws + 3 * BIG;       // enc attn-out / dec attn-out
  float* SC   = ws + 4 * BIG;       // scores (<= 64*256*256 floats)
  float* TMP  = SC + 4194304LL;     // 2048*512
  float* TMP2 = TMP + 1048576LL;    // 2048*512

  float* LAT = out_lat;  // latent lives in the output buffer

  const int   MROW   = PB * PNL;                  // 2048 latent rows
  const float inv512 = 1.0f / sqrtf((float)PD);   // 1/sqrt(512)
  const float inv32  = 1.0f / sqrtf((float)PDD);  // 1/sqrt(32)

  // latent = broadcast(latent_init)
  k_bcast<<<4096, 256, 0, stream>>>(lat_init, LAT, (long long)PNL * PD,
                                    (long long)PB * PNL * PD);

  // ---- cross attention (nh=1, nd=512), scores into `scores` buffer ----
  auto cross = [&](const float* qw, const float* qb, const float* kw,
                   const float* kb, const float* vw, const float* vb,
                   const float* ow, const float* ob, float* scores) {
    gemm(stream, false, LAT, qw, qb, TMP, MROW, PD, PD, PD, PD, PD,
         1, 0, 0, 0, 1, 0, 0, 0, 1.f, false);                       // Q
    gemm(stream, false, input, kw, kb, BUF0, PB * PS, PD, PD, PD, PD, PD,
         1, 0, 0, 0, 1, 0, 0, 0, 1.f, false);                       // K
    gemm(stream, false, input, vw, vb, BUF1, PB * PS, PD, PD, PD, PD, PD,
         1, 0, 0, 0, 1, 0, 0, 0, 1.f, false);                       // V
    gemm(stream, true, TMP, BUF0, nullptr, scores, PNL, PS, PD, PD, PD, PS,
         PB, (long long)PNL * PD, (long long)PS * PD, (long long)PNL * PS,
         1, 0, 0, 0, inv512, false);                                // Q K^T
    k_softmax<<<PB * PNL, 256, 0, stream>>>(scores, PS);
    gemm(stream, false, scores, BUF1, nullptr, TMP2, PNL, PD, PS, PS, PD, PD,
         PB, (long long)PNL * PS, (long long)PS * PD, (long long)PNL * PD,
         1, 0, 0, 0, 1.f, false);                                   // P V
    relu_ip(stream, TMP2, (long long)MROW * PD);
    gemm(stream, false, TMP2, ow, ob, LAT, MROW, PD, PD, PD, PD, PD,
         1, 0, 0, 0, 1, 0, 0, 0, 1.f, false);                       // O proj
  };

  // ---- encoder layer (nh=8, per-head nd=512, proj dim 4096) ----
  auto enc_layer = [&](int li) {
    const int  L   = LBASE + 16 * li;
    const float *f1b = F(L + 0), *f1w = F(L + 1), *f2b = F(L + 2),
                *f2w = F(L + 3), *ln1b = F(L + 4), *ln1g = F(L + 5),
                *ln2b = F(L + 6), *ln2g = F(L + 7), *mkb = F(L + 8),
                *mkw = F(L + 9), *mob = F(L + 10), *mow = F(L + 11),
                *mqb = F(L + 12), *mqw = F(L + 13), *mvb = F(L + 14),
                *mvw = F(L + 15);
    const int HD = PH * PD;  // 4096
    gemm(stream, false, LAT, mqw, mqb, BUF0, MROW, HD, PD, PD, HD, HD,
         1, 0, 0, 0, 1, 0, 0, 0, 1.f, false);
    gemm(stream, false, LAT, mkw, mkb, BUF1, MROW, HD, PD, PD, HD, HD,
         1, 0, 0, 0, 1, 0, 0, 0, 1.f, false);
    gemm(stream, false, LAT, mvw, mvb, BUF2, MROW, HD, PD, PD, HD, HD,
         1, 0, 0, 0, 1, 0, 0, 0, 1.f, false);
    // scores[b,h] = Q_bh K_bh^T / sqrt(512)  -> SC [B,H,256,256]
    gemm(stream, true, BUF0, BUF1, nullptr, SC, PNL, PNL, PD, HD, HD, PNL,
         PB, (long long)PNL * HD, (long long)PNL * HD,
         (long long)PH * PNL * PNL, PH, PD, PD, (long long)PNL * PNL,
         inv512, false);
    k_softmax<<<PB * PH * PNL, 256, 0, stream>>>(SC, PNL);
    // attn-out[b,t,h*512+d] = P_bh V_bh
    gemm(stream, false, SC, BUF2, nullptr, BUF3, PNL, PD, PNL, PNL, HD, HD,
         PB, (long long)PH * PNL * PNL, (long long)PNL * HD,
         (long long)PNL * HD, PH, (long long)PNL * PNL, PD, PD, 1.f, false);
    relu_ip(stream, BUF3, (long long)MROW * HD);
    gemm(stream, false, BUF3, mow, mob, TMP, MROW, PD, HD, HD, PD, PD,
         1, 0, 0, 0, 1, 0, 0, 0, 1.f, false);
    k_ln_res<<<MROW, 256, 0, stream>>>(TMP, LAT, ln1g, ln1b, PD);
    gemm(stream, false, LAT, f1w, f1b, TMP, MROW, PD, PD, PD, PD, PD,
         1, 0, 0, 0, 1, 0, 0, 0, 1.f, true);   // relu(ffn1)
    gemm(stream, false, TMP, f2w, f2b, TMP2, MROW, PD, PD, PD, PD, PD,
         1, 0, 0, 0, 1, 0, 0, 0, 1.f, false);
    k_ln_res<<<MROW, 256, 0, stream>>>(TMP2, LAT, ln2g, ln2b, PD);
  };

  cross(c1qw, c1qb, c1kw, c1kb, c1vw, c1vb, c1ow, c1ob, SC);
  for (int i = 0; i < 4; ++i) enc_layer(i);
  cross(c2qw, c2qb, c2kw, c2kb, c2vw, c2vb, c2ow, c2ob, out_attn);  // prob2 out
  for (int i = 0; i < 4; ++i) enc_layer(i);

  // sigma = linear(relu(max over latents))
  k_sigma<<<PB, 256, 0, stream>>>(LAT, sigw, sigb, out_sigma);

  // ---- decoder cross-attention (nh=1, nd=32) + color head ----
  gemm(stream, false, query, dqw, dqb, BUF0, PB * PQ, PDD, PDC, PDC, PDD, PDD,
       1, 0, 0, 0, 1, 0, 0, 0, 1.f, false);                         // q
  gemm(stream, false, LAT, dkw, dkb, BUF1, MROW, PDD, PD, PD, PDD, PDD,
       1, 0, 0, 0, 1, 0, 0, 0, 1.f, false);                         // k
  gemm(stream, false, LAT, dvw, dvb, BUF2, MROW, PDD, PD, PD, PDD, PDD,
       1, 0, 0, 0, 1, 0, 0, 0, 1.f, false);                         // v
  gemm(stream, true, BUF0, BUF1, nullptr, SC, PQ, PNL, PDD, PDD, PDD, PNL,
       PB, (long long)PQ * PDD, (long long)PNL * PDD, (long long)PQ * PNL,
       1, 0, 0, 0, inv32, false);                                   // q k^T
  k_softmax<<<PB * PQ, 256, 0, stream>>>(SC, PNL);
  gemm(stream, false, SC, BUF2, nullptr, BUF3, PQ, PDD, PNL, PNL, PDD, PDD,
       PB, (long long)PQ * PNL, (long long)PNL * PDD, (long long)PQ * PDD,
       1, 0, 0, 0, 1.f, false);                                     // P v
  relu_ip(stream, BUF3, (long long)PB * PQ * PDD);
  gemm(stream, false, BUF3, dow, dob, TMP, PB * PQ, PDD, PDD, PDD, PDD, PDD,
       1, 0, 0, 0, 1, 0, 0, 0, 1.f, false);                         // o proj
  relu_ip(stream, TMP, (long long)PB * PQ * PDD);
  k_color<<<(PB * PQ + 255) / 256, 256, 0, stream>>>(TMP, colw, colb, out_color);
}
